// MultiAttentionLayer_75960791597352
// MI455X (gfx1250) — compile-verified, matrix operand order checked by reading, not run on hardware
//
#include <hip/hip_runtime.h>
#include <hip/hip_bf16.h>

typedef __attribute__((ext_vector_type(16))) _Float16 v16h;
typedef __attribute__((ext_vector_type(8)))  float    v8f;
typedef unsigned int u32x4 __attribute__((ext_vector_type(4)));
typedef int          i32x8 __attribute__((ext_vector_type(8)));
typedef int          i32x4 __attribute__((ext_vector_type(4)));

#define BB   2
#define SS   2048
#define EE   1024
#define HH   16
#define DD   64
#define HD   1024      // H*D
#define MTOT 4096      // B*S

#define ROWS      32           // Sq rows per fused-attention workgroup
#define LDSTRIDE  (SS + 8)     // f32 stride, padded to break 64-bank conflicts

// TDM-staged A-tile in out_proj: 16 rows x 1024 halves, with the TDM pad
// feature inserting 1 DWORD (2 halves) after every 256 DWORDs (512 halves)
// -> row stride 1028 halves = 2056 B  (2056/4 % 64 == 2 banks -> conflict-free
// row-parallel fragment reads).
#define ATILE_STRIDE_H 1028

// ---------------------------------------------------------------------------
// Fragment loaders for V_WMMA_F32_16X16X32_F16 (wave32).
// 16-bit A-matrix 16x32 layout (ISA 7.12.2):
//   lanes 0-15 : row M=lane,    halves j=0..7 -> K=j,   j=8..15 -> K=16+(j-8)
//   lanes 16-31: row M=lane-16, halves j=0..7 -> K=8+j, j=8..15 -> K=24+(j-8)
// B fragment = same layout applied to B^T (lane indexes the N column).
// ---------------------------------------------------------------------------
__device__ __forceinline__ v16h ldfrag_h(const _Float16* rowptr, int half) {
  v16h r;
#pragma unroll
  for (int i = 0; i < 8; ++i) {
    r[i]     = rowptr[8 * half + i];
    r[i + 8] = rowptr[16 + 8 * half + i];
  }
  return r;
}

// f32 source (LDS scores), converted + scaled into an f16 fragment.
__device__ __forceinline__ v16h ldfrag_f32_scaled(const float* rowptr, int half,
                                                  float scale) {
  v16h r;
#pragma unroll
  for (int i = 0; i < 8; ++i) {
    r[i]     = (_Float16)(rowptr[8 * half + i] * scale);
    r[i + 8] = (_Float16)(rowptr[16 + 8 * half + i] * scale);
  }
  return r;
}

// A-fragment from the TDM-padded LDS tile (2 pad halves every 512 halves).
// Chunks (8 halves, k0 multiple of 32) never straddle a pad point.
__device__ __forceinline__ v16h ldfrag_lds_pad(const _Float16* aTile, int row,
                                               int k0, int half) {
  const _Float16* rp = aTile + row * ATILE_STRIDE_H;
  const int kA = k0 + 8 * half;
  const int kB = k0 + 16 + 8 * half;
  const int oA = kA + ((kA >> 9) << 1);
  const int oB = kB + ((kB >> 9) << 1);
  v16h r;
#pragma unroll
  for (int i = 0; i < 8; ++i) {
    r[i]     = rp[oA + i];
    r[i + 8] = rp[oB + i];
  }
  return r;
}

// ---------------------------------------------------------------------------
// Prepass: f32 -> f16 convert, and weight transpose+convert (W[K,N] -> WT[N,K])
// ---------------------------------------------------------------------------
__global__ void cvt_f32_to_f16(const float* __restrict__ src,
                               _Float16* __restrict__ dst, int n) {
  int i = blockIdx.x * blockDim.x + threadIdx.x;
  if (i < n) dst[i] = (_Float16)src[i];
}

__global__ void transpose_to_f16(const float* __restrict__ W,
                                 _Float16* __restrict__ WT, int K, int N) {
  int o = blockIdx.x * blockDim.x + threadIdx.x;
  if (o < K * N) {
    int n = o / K, k = o - n * K;
    WT[o] = (_Float16)W[(long)k * N + n];
  }
}

// ---------------------------------------------------------------------------
// QKV projection: [4096,1024] @ [1024,1024] per z in {q,k,v}.
// One 16x16 output tile per wave, K=1024 in steps of 32 (32 WMMAs).
// Q,K stored f16 [B,H,S,D]; V stored transposed f16 [B,H,D,S].
// ---------------------------------------------------------------------------
__global__ void qkv_proj(const _Float16* __restrict__ qh, const _Float16* __restrict__ kh,
                         const _Float16* __restrict__ vh,
                         const _Float16* __restrict__ wqT, const _Float16* __restrict__ wkT,
                         const _Float16* __restrict__ wvT,
                         const float* __restrict__ bq, const float* __restrict__ bk,
                         const float* __restrict__ bv,
                         _Float16* __restrict__ Qp, _Float16* __restrict__ Kp,
                         _Float16* __restrict__ Vt) {
  const int lane = threadIdx.x & 31;
  const int wave = threadIdx.x >> 5;
  const int half = lane >> 4;
  const int nl   = lane & 15;
  const int w     = blockIdx.x * 8 + wave;   // 16384 wave-tiles: 256 x 64
  const int tileM = w >> 6;
  const int tileN = w & 63;
  const int m0 = tileM * 16, n0 = tileN * 16;

  const _Float16* X; const _Float16* WT; const float* bias;
  const int z = blockIdx.z;
  if (z == 0)      { X = qh; WT = wqT; bias = bq; }
  else if (z == 1) { X = kh; WT = wkT; bias = bk; }
  else             { X = vh; WT = wvT; bias = bv; }

  v8f acc = {};
  for (int k0 = 0; k0 < EE; k0 += 32) {
    v16h a = ldfrag_h(X  + (long)(m0 + nl) * EE + k0, half);
    v16h b = ldfrag_h(WT + (long)(n0 + nl) * EE + k0, half);
    acc = __builtin_amdgcn_wmma_f32_16x16x32_f16(false, a, false, b,
                                                 (short)0, acc, false, false);
  }
  const int   n    = n0 + nl;
  const int   h    = n >> 6, d = n & 63;
  const float bval = bias[n];
#pragma unroll
  for (int r = 0; r < 8; ++r) {
    const int m  = m0 + r + 8 * half;
    const int b_ = m >> 11, s = m & 2047;
    const float val = acc[r] + bval;
    if (z == 0)
      Qp[(((long)(b_ * HH + h)) * SS + s) * DD + d] = (_Float16)val;
    else if (z == 1)
      Kp[(((long)(b_ * HH + h)) * SS + s) * DD + d] = (_Float16)val;
    else
      Vt[(((long)(b_ * HH + h)) * DD + d) * SS + s] = (_Float16)val;
  }
}

// ---------------------------------------------------------------------------
// Fused attention: per workgroup = one (b,h) and a 32-row Sq block.
//  Phase 1: scores = Q_blk @ K^T * 1/8 into LDS (32 x 2048 f32, padded stride)
//  Phase 2: softmax in LDS; normalized attn written once (coalesced,
//           NON-TEMPORAL: write-once data must not thrash the 192MB L2)
//  Phase 3: attn @ V from LDS (normalization folded into fragment convert)
// LDS use: 32*2056*4 = 257 KB of the WGP's 320 KB.
// ---------------------------------------------------------------------------
__global__ void fused_attention(const _Float16* __restrict__ Qp,
                                const _Float16* __restrict__ Kp,
                                const _Float16* __restrict__ Vt,
                                float* __restrict__ attn,
                                _Float16* __restrict__ ctx) {
  __shared__ float smem[ROWS * LDSTRIDE];     // exp/scores block
  __shared__ float part[ROWS][8];             // per-row partial reductions
  __shared__ float rowmaxs[ROWS];
  __shared__ float rowinv[ROWS];

  const int t    = threadIdx.x;
  const int lane = t & 31;
  const int wave = t >> 5;
  const int half = lane >> 4;
  const int nl   = lane & 15;

  const int sqb = blockIdx.x;                 // 64 Sq blocks
  const int bh  = blockIdx.y;                 // 32 (b,h) pairs
  const int sq0 = sqb * ROWS;

  const _Float16* Qb = Qp + (long)bh * SS * DD;
  const _Float16* Kb = Kp + (long)bh * SS * DD;
  const _Float16* Vb = Vt + (long)bh * DD * SS;

  // ---- Phase 1: score block into LDS --------------------------------------
  {
    const int tileM = wave >> 2;              // 0..1 (two 16-row halves)
    const int wsub  = wave & 3;               // 4 waves share one M tile
    const _Float16* qrow = Qb + (long)(sq0 + tileM * 16 + nl) * DD;
    const v16h a0 = ldfrag_h(qrow,      half);
    const v16h a1 = ldfrag_h(qrow + 32, half);

    for (int i = 0; i < 32; ++i) {
      const int tileN = wsub + 4 * i;         // covers 0..127
      const int n0 = tileN * 16;
      const _Float16* krow = Kb + (long)(n0 + nl) * DD;
      __builtin_prefetch(krow + 4 * 16 * DD, 0, 0);   // next tile for this wave
      v16h b0 = ldfrag_h(krow,      half);
      v16h b1 = ldfrag_h(krow + 32, half);
      v8f acc = {};
      acc = __builtin_amdgcn_wmma_f32_16x16x32_f16(false, a0, false, b0,
                                                   (short)0, acc, false, false);
      acc = __builtin_amdgcn_wmma_f32_16x16x32_f16(false, a1, false, b1,
                                                   (short)0, acc, false, false);
#pragma unroll
      for (int r = 0; r < 8; ++r) {
        const int rl = tileM * 16 + r + 8 * half;
        smem[rl * LDSTRIDE + n0 + nl] = acc[r] * 0.125f;   // 1/sqrt(64)
      }
    }
  }
  __syncthreads();

  // ---- Phase 2: softmax in LDS --------------------------------------------
  {
    const int rl  = t >> 3;                   // 32 rows, 8 threads per row
    const int sub = t & 7;
    float* rowp = smem + rl * LDSTRIDE;

    float mx = -3.0e38f;
    for (int c = sub * 256; c < sub * 256 + 256; ++c) mx = fmaxf(mx, rowp[c]);
    part[rl][sub] = mx;
    __syncthreads();
    if (sub == 0) {
      float m2 = part[rl][0];
#pragma unroll
      for (int j = 1; j < 8; ++j) m2 = fmaxf(m2, part[rl][j]);
      rowmaxs[rl] = m2;
    }
    __syncthreads();

    const float rm = rowmaxs[rl];
    float sm = 0.f;
    for (int c = sub * 256; c < sub * 256 + 256; ++c) {
      const float e = __expf(rowp[c] - rm);
      rowp[c] = e;
      sm += e;
    }
    part[rl][sub] = sm;
    __syncthreads();
    if (sub == 0) {
      float s2 = 0.f;
#pragma unroll
      for (int j = 0; j < 8; ++j) s2 += part[rl][j];
      rowinv[rl] = 1.0f / s2;
    }
    __syncthreads();
  }

  // ---- Phase 2b: single coalesced NON-TEMPORAL write of normalized attn ---
  {
    float* out = attn + ((long)bh * SS + sq0) * SS;
    for (int idx = t; idx < ROWS * SS; idx += 256) {
      const int row = idx >> 11;              // SS == 2048
      const int col = idx & (SS - 1);
      const float val = smem[row * LDSTRIDE + col] * rowinv[row];
      __builtin_nontemporal_store(val, &out[(long)row * SS + col]);
    }
  }
  __syncthreads();

  // ---- Phase 3: attn @ V from LDS -----------------------------------------
  {
    const int tileM = wave >> 2;              // 0..1
    const int tileN = wave & 3;               // 0..3  (D = 64 -> 4 tiles)
    const int mrow  = tileM * 16 + nl;        // this lane's attn row
    const float inv = rowinv[mrow];
    const float* arow = smem + mrow * LDSTRIDE;
    const _Float16* vrow = Vb + (long)(tileN * 16 + nl) * SS;

    v8f acc = {};
    for (int k0 = 0; k0 < SS; k0 += 32) {
      __builtin_prefetch(vrow + k0 + 128, 0, 0);
      v16h a = ldfrag_f32_scaled(arow + k0, half, inv);
      v16h b = ldfrag_h(vrow + k0, half);
      acc = __builtin_amdgcn_wmma_f32_16x16x32_f16(false, a, false, b,
                                                   (short)0, acc, false, false);
    }
    const int b_ = bh >> 4, h = bh & 15;
#pragma unroll
    for (int r = 0; r < 8; ++r) {
      const int s = sq0 + tileM * 16 + r + 8 * half;
      const int d = tileN * 16 + nl;
      ctx[((long)(b_ * SS + s)) * HD + h * DD + d] = (_Float16)acc[r];
    }
  }
}

// ---------------------------------------------------------------------------
// Output projection + bias + residual -> xbuf (f32).
// All 8 waves of a block share the same A tile (ctx rows m0..m0+15), so one
// TENSOR_LOAD_TO_LDS (TDM) stages it into LDS once per block, with the TDM
// pad feature de-conflicting the LDS banks. B (woT) streams from L2.
// ---------------------------------------------------------------------------
__global__ void out_proj(const _Float16* __restrict__ ctx,
                         const _Float16* __restrict__ woT,
                         const float* __restrict__ bo,
                         const float* __restrict__ qres,
                         float* __restrict__ xbuf) {
  __shared__ __attribute__((aligned(16))) _Float16 aTile[16 * ATILE_STRIDE_H];

  const int lane = threadIdx.x & 31;
  const int wave = threadIdx.x >> 5;
  const int half = lane >> 4;
  const int nl   = lane & 15;
  const int w     = blockIdx.x * 8 + wave;   // 16384 wave-tiles: 256 x 64
  const int tileN = w & 63;
  const int n0 = tileN * 16;
  // tileM is uniform across the block: 8 consecutive w never cross a /64 edge.
  const int m0 = (blockIdx.x >> 3) * 16;

  // ---- TDM: stage A tile (16 x 1024 f16 rows of ctx) into LDS -------------
  if (wave == 0) {
    const unsigned long long gaddr =
        (unsigned long long)(uintptr_t)(ctx + (long)m0 * HD);
    const unsigned int lds_off = (unsigned int)(uintptr_t)aTile; // low 32b = LDS byte offset

    u32x4 g0;
    g0[0] = 1u;                                     // count=1 user descriptor
    g0[1] = lds_off;                                // lds_addr (bytes)
    g0[2] = (unsigned int)(gaddr & 0xFFFFFFFFu);    // global_addr[31:0]
    g0[3] = (unsigned int)((gaddr >> 32) & 0x01FFFFFFu) | (2u << 30); // [56:32]+type=2

    i32x8 g1;
    g1[0] = (1 << 16)        // data_size = 2 bytes (f16)
          | (1 << 20)        // pad_enable
          | (7 << 22)        // pad_interval: 256 DWORDs
          | (0 << 25);       // pad_amount: 1 DWORD
    g1[1] = (int)(0x0400u << 16);   // tensor_dim0 = 1024 (bits 79:48, low16)
    g1[2] = (int)(0x1000u << 16);   // tensor_dim1 = 4096 (bits 111:80, low16)
    g1[3] = (int)(0x0400u << 16);   // tile_dim0  = 1024 (bits 127:112)
    g1[4] = 16;                     // tile_dim1 = 16, tile_dim2 = 0
    g1[5] = 1024;                   // tensor_dim0_stride[31:0]
    g1[6] = 0;
    g1[7] = 0;

    i32x4 g2 = {0, 0, 0, 0};        // dims 2/3 unused (tile_dim3 = 0)
    i32x4 g3 = {0, 0, 0, 0};

#if defined(__clang_major__) && __clang_major__ >= 23
    i32x8 gz = {0, 0, 0, 0, 0, 0, 0, 0};
    __builtin_amdgcn_tensor_load_to_lds(g0, g1, g2, g3, gz, 0);
#else
    __builtin_amdgcn_tensor_load_to_lds(g0, g1, g2, g3, 0);
#endif
    __builtin_amdgcn_s_wait_tensorcnt(0);
  }
  __syncthreads();

  v8f acc = {};
  for (int k0 = 0; k0 < HD; k0 += 32) {
    v16h a = ldfrag_lds_pad(aTile, nl, k0, half);
    v16h b = ldfrag_h(woT + (long)(n0 + nl) * HD + k0, half);
    acc = __builtin_amdgcn_wmma_f32_16x16x32_f16(false, a, false, b,
                                                 (short)0, acc, false, false);
  }
  const int   n    = n0 + nl;
  const float bval = bo[n];
#pragma unroll
  for (int r = 0; r < 8; ++r) {
    const int m = m0 + r + 8 * half;
    xbuf[(long)m * EE + n] = acc[r] + bval + qres[(long)m * EE + n];
  }
}

// ---------------------------------------------------------------------------
// LayerNorm per row of xbuf -> y. One 256-thread block per row of 1024.
// var = E[x^2] - mu^2 (matches jnp.var, biased).
// ---------------------------------------------------------------------------
__global__ void layernorm_rows(const float* __restrict__ xbuf,
                               const float* __restrict__ gamma,
                               const float* __restrict__ beta,
                               float* __restrict__ y) {
  const long row = blockIdx.x;               // 4096 rows
  const float* p = xbuf + row * EE;
  const int t = threadIdx.x;

  float x[4];
  float s = 0.f, s2 = 0.f;
#pragma unroll
  for (int i = 0; i < 4; ++i) {
    x[i] = p[t + i * 256];
    s  += x[i];
    s2 += x[i] * x[i];
  }
  __shared__ float rs[256];
  __shared__ float rs2[256];
  rs[t] = s; rs2[t] = s2;
  __syncthreads();
  for (int o = 128; o > 0; o >>= 1) {
    if (t < o) { rs[t] += rs[t + o]; rs2[t] += rs2[t + o]; }
    __syncthreads();
  }
  const float mu  = rs[0] * (1.0f / EE);
  const float var = rs2[0] * (1.0f / EE) - mu * mu;
  const float inv = rsqrtf(var + 1e-5f);
  float* out = y + row * EE;
#pragma unroll
  for (int i = 0; i < 4; ++i) {
    const int n = t + i * 256;
    out[n] = (x[i] - mu) * inv * gamma[n] + beta[n];
  }
}

// ---------------------------------------------------------------------------
// Host launcher
// ---------------------------------------------------------------------------
extern "C" void kernel_launch(void* const* d_in, const int* in_sizes, int n_in,
                              void* d_out, int out_size, void* d_ws, size_t ws_size,
                              hipStream_t stream) {
  (void)in_sizes; (void)n_in; (void)out_size; (void)ws_size;

  const float* q     = (const float*)d_in[0];
  const float* k     = (const float*)d_in[1];
  const float* v     = (const float*)d_in[2];
  const float* Wq    = (const float*)d_in[3];
  const float* bq    = (const float*)d_in[4];
  const float* Wk    = (const float*)d_in[5];
  const float* bk    = (const float*)d_in[6];
  const float* Wv    = (const float*)d_in[7];
  const float* bv    = (const float*)d_in[8];
  const float* Wo    = (const float*)d_in[9];
  const float* bo    = (const float*)d_in[10];
  const float* gamma = (const float*)d_in[11];
  const float* beta  = (const float*)d_in[12];

  float* y    = (float*)d_out;                              // [B,S,E]
  float* attn = (float*)d_out + (size_t)BB * SS * EE;       // [B,H,S,S]

  const size_t MB = (size_t)1 << 20;
  char* ws = (char*)d_ws;
  _Float16* qh   = (_Float16*)(ws +  0 * MB);   // [4096,1024] f16
  _Float16* kh   = (_Float16*)(ws +  8 * MB);
  _Float16* vh   = (_Float16*)(ws + 16 * MB);
  _Float16* wqT  = (_Float16*)(ws + 24 * MB);   // [1024,1024] f16 (N-major)
  _Float16* wkT  = (_Float16*)(ws + 26 * MB);
  _Float16* wvT  = (_Float16*)(ws + 28 * MB);
  _Float16* woT  = (_Float16*)(ws + 30 * MB);
  _Float16* Qp   = (_Float16*)(ws + 32 * MB);   // [B,H,S,D] f16
  _Float16* Kp   = (_Float16*)(ws + 40 * MB);   // [B,H,S,D] f16
  _Float16* Vt   = (_Float16*)(ws + 48 * MB);   // [B,H,D,S] f16
  _Float16* ctx  = (_Float16*)(ws + 56 * MB);   // [4096,1024] f16
  float*    xbuf = (float*)   (ws + 64 * MB);   // [4096,1024] f32

  const int NEL = MTOT * EE;                    // 4194304

  // Prepass: converts + weight transposes
  cvt_f32_to_f16<<<NEL / 256, 256, 0, stream>>>(q, qh, NEL);
  cvt_f32_to_f16<<<NEL / 256, 256, 0, stream>>>(k, kh, NEL);
  cvt_f32_to_f16<<<NEL / 256, 256, 0, stream>>>(v, vh, NEL);
  transpose_to_f16<<<(EE * HD) / 256, 256, 0, stream>>>(Wq, wqT, EE, HD);
  transpose_to_f16<<<(EE * HD) / 256, 256, 0, stream>>>(Wk, wkT, EE, HD);
  transpose_to_f16<<<(EE * HD) / 256, 256, 0, stream>>>(Wv, wvT, EE, HD);
  transpose_to_f16<<<(HD * EE) / 256, 256, 0, stream>>>(Wo, woT, HD, EE);

  // QKV projections (WMMA)
  dim3 gq(2048, 1, 3);
  qkv_proj<<<gq, 256, 0, stream>>>(qh, kh, vh, wqT, wkT, wvT, bq, bk, bv,
                                   Qp, Kp, Vt);

  // Fused scores + softmax + attn@V (WMMA, 257 KB LDS per workgroup)
  dim3 ga(SS / ROWS, BB * HH);                  // (64, 32)
  fused_attention<<<ga, 256, 0, stream>>>(Qp, Kp, Vt, attn, ctx);

  // Output projection + bias + residual (WMMA + TDM-staged A tile)
  out_proj<<<2048, 256, 0, stream>>>(ctx, woT, bo, q, xbuf);

  // LayerNorm -> y
  layernorm_rows<<<MTOT, 256, 0, stream>>>(xbuf, gamma, beta, y);
}
